// MultiHeadAttention_53154515256172
// MI455X (gfx1250) — matrix operand order repaired and checked by reading, not compiled, not run on hardware
//
#include <hip/hip_runtime.h>
#include <hip/hip_bf16.h>
#include <stdint.h>

// ---------------------------------------------------------------------------
// MHA forward on gfx1250 (wave32): WMMA bf16->f32, all tiles staged by TDM,
// DPP cross-lane softmax reductions.
// ---------------------------------------------------------------------------

typedef __attribute__((ext_vector_type(16))) __bf16 v16bf;
typedef __attribute__((ext_vector_type(8)))  float  v8f;
typedef __attribute__((ext_vector_type(4)))  unsigned int u32x4;
typedef __attribute__((ext_vector_type(8)))  int i32x8;
typedef __attribute__((ext_vector_type(4)))  int i32x4;

#define EMB   1024
#define DKDIM 1024
#define NH    16
#define HD    64
#define BATCH 4
#define SEQ   2048
#define MROWS (BATCH * SEQ)   // 8192

#if defined(__AMDGCN__) && __has_builtin(__builtin_amdgcn_tensor_load_to_lds)
#define HAVE_TDM 1
#else
#define HAVE_TDM 0
#endif

union AFrag { uint32_t u[8]; v16bf v; };

__device__ __forceinline__ uint16_t f2bf(float f) {
  uint32_t u = __float_as_uint(f);
  u += 0x7FFFu + ((u >> 16) & 1u);   // round-to-nearest-even
  return (uint16_t)(u >> 16);
}

__device__ __forceinline__ v8f vzero8() {
  v8f z = {0.f, 0.f, 0.f, 0.f, 0.f, 0.f, 0.f, 0.f};
  return z;
}

// ---- 16-lane (row) reductions via DPP: xor1, xor2, half-mirror, mirror ----
#if defined(__AMDGCN__) && __has_builtin(__builtin_amdgcn_update_dpp)
#define DPP_STEP_MAX(x, ctrl)                                                  \
  x = fmaxf(x, __int_as_float(__builtin_amdgcn_update_dpp(                     \
          __float_as_int(x), __float_as_int(x), ctrl, 0xF, 0xF, true)))
#define DPP_STEP_ADD(x, ctrl)                                                  \
  x = x + __int_as_float(__builtin_amdgcn_update_dpp(                          \
          __float_as_int(x), __float_as_int(x), ctrl, 0xF, 0xF, true))
__device__ __forceinline__ float red_max16(float x) {
  DPP_STEP_MAX(x, 0xB1);   // quad_perm(1,0,3,2)
  DPP_STEP_MAX(x, 0x4E);   // quad_perm(2,3,0,1)
  DPP_STEP_MAX(x, 0x141);  // ROW_HALF_MIRROR
  DPP_STEP_MAX(x, 0x140);  // ROW_MIRROR
  return x;
}
__device__ __forceinline__ float red_sum16(float x) {
  DPP_STEP_ADD(x, 0xB1);
  DPP_STEP_ADD(x, 0x4E);
  DPP_STEP_ADD(x, 0x141);
  DPP_STEP_ADD(x, 0x140);
  return x;
}
#else
__device__ __forceinline__ float red_max16(float x) {
  for (int off = 1; off < 16; off <<= 1) x = fmaxf(x, __shfl_xor(x, off, 32));
  return x;
}
__device__ __forceinline__ float red_sum16(float x) {
  for (int off = 1; off < 16; off <<= 1) x += __shfl_xor(x, off, 32);
  return x;
}
#endif

#if HAVE_TDM
__device__ __forceinline__ void wait_tensorcnt0() {
#if __has_builtin(__builtin_amdgcn_s_wait_tensorcnt)
  __builtin_amdgcn_s_wait_tensorcnt(0);
#else
  asm volatile("s_wait_tensorcnt 0x0" ::: "memory");
#endif
}

// 2D tile load via Tensor Data Mover (D# per CDNA5 ISA ch.8):
// tile_d0 contiguous 2B elements per row, tile_d1 rows, row pitch stride_elems.
// One issue per wave; EXEC ignored; tracked on TENSORcnt.
__device__ __forceinline__ void tdm_load_2d(uint32_t lds_off, const void* gptr,
                                            uint32_t tile_d0, uint32_t tile_d1,
                                            uint32_t stride_elems) {
  uint64_t ga = (uint64_t)(uintptr_t)gptr;
  // group0: count=1 | lds_addr | global_addr[56:0] | type=2
  u32x4 g0 = { 1u,
               lds_off,
               (uint32_t)ga,
               (uint32_t)((ga >> 32) & 0x01FFFFFFu) | (2u << 30) };
  // group1: data_size=2B; huge tensor dims (no OOB clip); tile dims; dim0 stride
  const uint32_t td0 = 1u << 20, td1 = 1u << 20;
  i32x8 g1 = { (int)(1u << 16),
               (int)((td0 & 0xFFFFu) << 16),
               (int)(((td0 >> 16) & 0xFFFFu) | ((td1 & 0xFFFFu) << 16)),
               (int)(((td1 >> 16) & 0xFFFFu) | ((tile_d0 & 0xFFFFu) << 16)),
               (int)(tile_d1 & 0xFFFFu),
               (int)stride_elems,
               0, 0 };
  i32x4 gz = {0, 0, 0, 0};
#if __clang_major__ >= 23
  i32x8 gz8 = {0, 0, 0, 0, 0, 0, 0, 0};
  __builtin_amdgcn_tensor_load_to_lds(g0, g1, gz, gz, gz8, 0);
#else
  __builtin_amdgcn_tensor_load_to_lds(g0, g1, gz, gz, 0);
#endif
}
#endif  // HAVE_TDM

// ---------------------------------------------------------------------------
// fp32 -> bf16 conversion, 4 elements / thread
// ---------------------------------------------------------------------------
__global__ void cvt_f32_bf16(const float* __restrict__ src,
                             uint16_t* __restrict__ dst, int n) {
  int i = (blockIdx.x * blockDim.x + threadIdx.x) * 4;
  if (i + 3 < n) {
    float4 f = *(const float4*)(src + i);
    uint32_t lo = (uint32_t)f2bf(f.x) | ((uint32_t)f2bf(f.y) << 16);
    uint32_t hi = (uint32_t)f2bf(f.z) | ((uint32_t)f2bf(f.w) << 16);
    uint2 p; p.x = lo; p.y = hi;
    *(uint2*)(dst + i) = p;
  }
}

// ---------------------------------------------------------------------------
// WMMA GEMM: C[M,N] = A[M,K] * B[K,N] (+bias), A/B bf16 row-major.
// 256 threads = 8 waves; block tile 64(M) x 256(N); wave tile 32x64.
// K-step 64; A and B tiles both DMA'd to LDS by the TDM (natural layout).
// ---------------------------------------------------------------------------
template <bool OUT_BF16>
__global__ __launch_bounds__(256) void gemm_bf16_wmma(
    const uint16_t* __restrict__ A, const uint16_t* __restrict__ Bm,
    void* __restrict__ C, const float* __restrict__ bias,
    int M, int N, int K) {
  __shared__ uint16_t Alds[64 * 64];     // [row][k]
  __shared__ uint16_t Blds[64 * 256];    // [k][n] (natural)

  const int tid = threadIdx.x;
  const int lane = tid & 31;
  const int wid  = tid >> 5;
  const int half = lane >> 4;
  const int l15  = lane & 15;
  const int wm = wid & 1;                // 2 waves along M
  const int wn = wid >> 1;               // 4 waves along N
  const int m0 = blockIdx.y * 64;
  const int n0 = blockIdx.x * 256;
#if HAVE_TDM
  const uint32_t alds_off = (uint32_t)(uintptr_t)(void*)Alds;
  const uint32_t blds_off = (uint32_t)(uintptr_t)(void*)Blds;
#endif

  v8f acc[2][4];
#pragma unroll
  for (int i = 0; i < 2; ++i)
#pragma unroll
    for (int j = 0; j < 4; ++j) acc[i][j] = vzero8();

  for (int k0 = 0; k0 < K; k0 += 64) {
    __syncthreads();
#if HAVE_TDM
    if (wid == 0) {
      tdm_load_2d(alds_off, A + (size_t)m0 * K + k0, /*d0*/64,  /*d1*/64, K);
      tdm_load_2d(blds_off, Bm + (size_t)k0 * N + n0, /*d0*/256, /*d1*/64, N);
      wait_tensorcnt0();
    }
#else
    {
      uint32_t* Al = (uint32_t*)Alds;
      for (int i = tid; i < 2048; i += 256) {
        int row = i >> 5;
        int kk  = (i & 31) * 2;
        Al[i] = *(const uint32_t*)(A + (size_t)(m0 + row) * K + k0 + kk);
      }
      uint32_t* Bl = (uint32_t*)Blds;
      for (int i = tid; i < 8192; i += 256) {
        int row = i >> 7;
        int nn  = (i & 127) * 2;
        Bl[i] = *(const uint32_t*)(Bm + (size_t)(k0 + row) * N + n0 + nn);
      }
    }
#endif
    if (k0 + 64 < K)
      __builtin_prefetch(Bm + (size_t)(k0 + 64 + (tid >> 3)) * N + n0 + ((tid & 7) << 5), 0, 1);
    __syncthreads();

#pragma unroll
    for (int ks = 0; ks < 64; ks += 32) {
      AFrag af[2];
#pragma unroll
      for (int mi = 0; mi < 2; ++mi) {
        int row = wm * 32 + mi * 16 + l15;
#pragma unroll
        for (int j = 0; j < 8; ++j) {
          int k = ks + 2 * j + 16 * half;  // contiguous k = 16*half + 2j (same order as B)
          af[mi].u[j] = *(const uint32_t*)&Alds[row * 64 + k];
        }
      }
      AFrag bfr[4];
#pragma unroll
      for (int ni = 0; ni < 4; ++ni) {
        int col = wn * 64 + ni * 16 + l15;
        int kb  = ks + 16 * half;                              // contiguous k = 16*half + 2j (same order as A)
#pragma unroll
        for (int j = 0; j < 8; ++j) {
          uint32_t lo = Blds[(kb + 2 * j) * 256 + col];
          uint32_t hi = Blds[(kb + 2 * j + 1) * 256 + col];
          bfr[ni].u[j] = lo | (hi << 16);
        }
      }
#pragma unroll
      for (int mi = 0; mi < 2; ++mi)
#pragma unroll
        for (int ni = 0; ni < 4; ++ni)
          acc[mi][ni] = __builtin_amdgcn_wmma_f32_16x16x32_bf16(
              false, af[mi].v, false, bfr[ni].v, (short)0, acc[mi][ni], false, false);
    }
  }

  // store (C/D layout: N = lane[3:0], M = vgpr + 8*lane[4])
#pragma unroll
  for (int mi = 0; mi < 2; ++mi)
#pragma unroll
    for (int ni = 0; ni < 4; ++ni) {
      int n = n0 + wn * 64 + ni * 16 + l15;
      float bv = bias ? bias[n] : 0.0f;
#pragma unroll
      for (int r = 0; r < 8; ++r) {
        int m = m0 + wm * 32 + mi * 16 + r + 8 * half;
        float val = acc[mi][ni][r] + bv;
        if (OUT_BF16) ((uint16_t*)C)[(size_t)m * N + n] = f2bf(val);
        else          ((float*)C)[(size_t)m * N + n] = val;
      }
    }
}

// ---------------------------------------------------------------------------
// Online-softmax step for one 16q x 32key S tile (two 16x16 D-frags).
// MASKED=false on interior (fully-causal-valid) tiles: no cmp/cndmask chain.
// ---------------------------------------------------------------------------
template <bool MASKED>
__device__ __forceinline__ void softmax_step(
    const v8f& s0v, const v8f& s1v, v8f (&acc)[4],
    float (&mrow)[8], float (&lrow)[8], uint16_t* Pw,
    int kb0, int q0w, int l15, int half) {
#pragma unroll
  for (int r = 0; r < 8; ++r) {
    int m = r + 8 * half;
    int qg = q0w + m;
    float s0 = s0v[r] * 0.125f;          // 1/sqrt(64)
    float s1 = s1v[r] * 0.125f;
    if (MASKED) {
      if (kb0 + l15 > qg)      s0 = -1e30f;
      if (kb0 + 16 + l15 > qg) s1 = -1e30f;
    }
    float tm = red_max16(fmaxf(s0, s1));
    float newm = fmaxf(mrow[r], tm);
    float alpha = __expf(mrow[r] - newm);
    float p0 = __expf(s0 - newm);
    float p1 = __expf(s1 - newm);
    float rs = red_sum16(p0 + p1);
    lrow[r] = lrow[r] * alpha + rs;
    mrow[r] = newm;
#pragma unroll
    for (int f = 0; f < 4; ++f) acc[f][r] *= alpha;
    Pw[m * 32 + l15]      = f2bf(p0);
    Pw[m * 32 + 16 + l15] = f2bf(p1);
  }
}

// ---------------------------------------------------------------------------
// Flash attention: grid (SEQ/64, BATCH*NH), 128 threads = 4 waves.
// Each wave: 16 q-rows, full 16x64 f32 context accumulator (4 D-frags).
// K and V tiles both DMA'd to LDS by the TDM (natural [key][d] layout).
// ---------------------------------------------------------------------------
__global__ __launch_bounds__(128) void attn_wmma(
    const uint16_t* __restrict__ Q, const uint16_t* __restrict__ Kmat,
    const uint16_t* __restrict__ V, uint16_t* __restrict__ CTX) {
  __shared__ uint16_t Klds[32 * 64];       // [key][d]
  __shared__ uint16_t Vlds[32 * 64];       // [key][d]
  __shared__ uint16_t Plds[4 * 16 * 32];   // per-wave P tile [16 q][32 key]

  const int tid  = threadIdx.x;
  const int lane = tid & 31;
  const int w    = tid >> 5;
  const int half = lane >> 4;
  const int l15  = lane & 15;
  const int bh = blockIdx.y;
  const int b = bh >> 4, h = bh & 15;
  const int q0w = blockIdx.x * 64 + w * 16;
  const size_t rowbase = (size_t)b * SEQ;
#if HAVE_TDM
  const uint32_t klds_off = (uint32_t)(uintptr_t)(void*)Klds;
  const uint32_t vlds_off = (uint32_t)(uintptr_t)(void*)Vlds;
#endif
  uint16_t* Pw = Plds + w * 512;

  // Q fragments (registers for whole kernel): 16 x 64, two contraction steps
  AFrag qf[2];
  {
    const uint16_t* qrow = Q + (rowbase + q0w + l15) * DKDIM + h * HD;
#pragma unroll
    for (int s = 0; s < 2; ++s)
#pragma unroll
      for (int j = 0; j < 8; ++j) {
        int d = s * 32 + 2 * j + 16 * half;
        qf[s].u[j] = *(const uint32_t*)(qrow + d);
      }
  }

  v8f acc[4];
#pragma unroll
  for (int f = 0; f < 4; ++f) acc[f] = vzero8();
  float mrow[8], lrow[8];
#pragma unroll
  for (int r = 0; r < 8; ++r) { mrow[r] = -1e30f; lrow[r] = 0.0f; }

  const int nkt = 2 * blockIdx.x + 2;   // 32-key tiles covering causal span

  for (int kt = 0; kt < nkt; ++kt) {
    const int kb0 = kt * 32;
    __syncthreads();   // previous iteration's K/V reads complete
    const uint16_t* Kg = Kmat + (rowbase + kb0) * DKDIM + h * HD;
    const uint16_t* Vg = V    + (rowbase + kb0) * DKDIM + h * HD;
#if HAVE_TDM
    if (w == 0) {
      tdm_load_2d(klds_off, Kg, /*d0*/HD, /*d1*/32, DKDIM);
      tdm_load_2d(vlds_off, Vg, /*d0*/HD, /*d1*/32, DKDIM);
      wait_tensorcnt0();
    }
#else
    {
      uint32_t* Kl = (uint32_t*)Klds;
      uint32_t* Vl = (uint32_t*)Vlds;
      for (int i = tid; i < 1024; i += 128) {
        int key = i >> 5;
        int d   = (i & 31) * 2;
        Kl[i] = *(const uint32_t*)(Kg + (size_t)key * DKDIM + d);
        Vl[i] = *(const uint32_t*)(Vg + (size_t)key * DKDIM + d);
      }
    }
#endif
    __syncthreads();   // staging complete

    // wave-uniform causal skip: tile entirely above this wave's rows
    const bool active = (kb0 <= q0w + 15);
    if (active) {
      // S = Q * K^T : two 16-key subtiles, contraction over d (2 WMMA each)
      v8f sfrag[2];
#pragma unroll
      for (int sub = 0; sub < 2; ++sub) {
        v8f s = vzero8();
#pragma unroll
        for (int st = 0; st < 2; ++st) {
          AFrag kf;
          int col = sub * 16 + l15;
          int kb  = st * 32 + 16 * half;
#pragma unroll
          for (int j = 0; j < 8; ++j)
            kf.u[j] = *(const uint32_t*)&Klds[col * 64 + kb + 2 * j];
          s = __builtin_amdgcn_wmma_f32_16x16x32_bf16(
              false, qf[st].v, false, kf.v, (short)0, s, false, false);
        }
        sfrag[sub] = s;
      }
      if (kb0 + 31 > q0w)   // diagonal-crossing tile: needs causal mask
        softmax_step<true >(sfrag[0], sfrag[1], acc, mrow, lrow, Pw, kb0, q0w, l15, half);
      else                  // interior tile: no mask compares at all
        softmax_step<false>(sfrag[0], sfrag[1], acc, mrow, lrow, Pw, kb0, q0w, l15, half);
    }
    __syncthreads();   // P published (workgroup barrier, all waves)

    if (active) {
      // ctx += P(16x32) * V(32x64); V frag gathered from natural layout
      AFrag pf;
#pragma unroll
      for (int j = 0; j < 8; ++j) {
        int k = 2 * j + 16 * half;
        pf.u[j] = *(const uint32_t*)&Pw[l15 * 32 + k];
      }
#pragma unroll
      for (int f = 0; f < 4; ++f) {
        AFrag vf;
        int col = f * 16 + l15;
        int kb  = 16 * half;
#pragma unroll
        for (int j = 0; j < 8; ++j) {
          uint32_t lo = Vlds[(kb + 2 * j) * 64 + col];
          uint32_t hi = Vlds[(kb + 2 * j + 1) * 64 + col];
          vf.u[j] = lo | (hi << 16);
        }
        acc[f] = __builtin_amdgcn_wmma_f32_16x16x32_bf16(
            false, pf.v, false, vf.v, (short)0, acc[f], false, false);
      }
    }
  }

  // normalize and store ctx (bf16, [B*T, DK] with per-head column offset)
#pragma unroll
  for (int f = 0; f < 4; ++f) {
    int d = h * HD + f * 16 + l15;
#pragma unroll
    for (int r = 0; r < 8; ++r) {
      int m = r + 8 * half;
      float val = acc[f][r] / lrow[r];
      CTX[(rowbase + q0w + m) * DKDIM + d] = f2bf(val);
    }
  }
}

// ---------------------------------------------------------------------------
// Host launcher
// ---------------------------------------------------------------------------
extern "C" void kernel_launch(void* const* d_in, const int* in_sizes, int n_in,
                              void* d_out, int out_size, void* d_ws, size_t ws_size,
                              hipStream_t stream) {
  const float* x  = (const float*)d_in[0];
  const float* Wq = (const float*)d_in[1];
  const float* Wk = (const float*)d_in[2];
  const float* Wv = (const float*)d_in[3];
  const float* Wo = (const float*)d_in[4];
  const float* bo = (const float*)d_in[5];
  float* out = (float*)d_out;

  // workspace layout (bf16), ~88 MB total
  uint16_t* Xb  = (uint16_t*)d_ws;
  uint16_t* Wqb = Xb  + (size_t)MROWS * EMB;
  uint16_t* Wkb = Wqb + (size_t)EMB * DKDIM;
  uint16_t* Wvb = Wkb + (size_t)EMB * DKDIM;
  uint16_t* Wob = Wvb + (size_t)EMB * DKDIM;
  uint16_t* Qb  = Wob + (size_t)DKDIM * EMB;
  uint16_t* Kb  = Qb  + (size_t)MROWS * DKDIM;
  uint16_t* Vb  = Kb  + (size_t)MROWS * DKDIM;
  uint16_t* Cb  = Vb  + (size_t)MROWS * DKDIM;

  // fp32 -> bf16 conversions
  {
    int nx = MROWS * EMB;                 // 8,388,608
    cvt_f32_bf16<<<nx / 4 / 256, 256, 0, stream>>>(x, Xb, nx);
    int nw = EMB * DKDIM;                 // 1,048,576
    cvt_f32_bf16<<<nw / 4 / 256, 256, 0, stream>>>(Wq, Wqb, nw);
    cvt_f32_bf16<<<nw / 4 / 256, 256, 0, stream>>>(Wk, Wkb, nw);
    cvt_f32_bf16<<<nw / 4 / 256, 256, 0, stream>>>(Wv, Wvb, nw);
    cvt_f32_bf16<<<nw / 4 / 256, 256, 0, stream>>>(Wo, Wob, nw);
  }

  // Q/K/V projections (bf16 out)
  dim3 ggrid(DKDIM / 256, MROWS / 64);
  gemm_bf16_wmma<true><<<ggrid, 256, 0, stream>>>(Xb, Wqb, Qb, nullptr, MROWS, DKDIM, EMB);
  gemm_bf16_wmma<true><<<ggrid, 256, 0, stream>>>(Xb, Wkb, Kb, nullptr, MROWS, DKDIM, EMB);
  gemm_bf16_wmma<true><<<ggrid, 256, 0, stream>>>(Xb, Wvb, Vb, nullptr, MROWS, DKDIM, EMB);

  // causal flash attention
  attn_wmma<<<dim3(SEQ / 64, BATCH * NH), 128, 0, stream>>>(Qb, Kb, Vb, Cb);

  // output projection with bias (f32 out)
  gemm_bf16_wmma<false><<<dim3(EMB / 256, MROWS / 64), 256, 0, stream>>>(
      Cb, Wob, out, bo, MROWS, EMB, DKDIM);
}
